// multihead_self_attention_78108275245090
// MI455X (gfx1250) — compile-verified, matrix operand order checked by reading, not run on hardware
//
#include <hip/hip_runtime.h>
#include <hip/hip_bf16.h>

#define D_MODEL   1024
#define NUM_HEADS 16
#define DK        64
#define BATCH     2
#define SEQ       2048
#define BS        (BATCH * SEQ)

typedef __attribute__((ext_vector_type(16))) __bf16 v16bf;
typedef __attribute__((ext_vector_type(8)))  __bf16 v8bf;
typedef __attribute__((ext_vector_type(8)))  float  v8f;

// ---------------------------------------------------------------------------
// Fragment loader for 16-bit A fragments (16xK rows striped in a lane) and for
// B fragments when memory holds B^T row-major (identical lane/K pattern).
// Per ISA 7.12.2: lane L (L<16) element e -> K = (e&7) + 16*(e>>3);
// lanes 16..31 add +8. Two contiguous 16B loads per lane (at +0 and +16
// elements, i.e. 32B apart).
// ---------------------------------------------------------------------------
__device__ __forceinline__ v16bf load_frag16(const __bf16* __restrict__ base,
                                             int ld, int lane) {
  const int r  = lane & 15;
  const int kb = (lane >> 4) << 3;          // 0 or 8
  const __bf16* p = base + (size_t)r * ld + kb;
  v8bf lo = *(const v8bf*)(p);              // K = kb .. kb+7   -> e0..e7
  v8bf hi = *(const v8bf*)(p + 16);         // K = kb+16..kb+23 -> e8..e15
  v16bf f;
#pragma unroll
  for (int i = 0; i < 8; ++i) { f[i] = lo[i]; f[i + 8] = hi[i]; }
  return f;
}

__device__ __forceinline__ v8f wmma_bf16(v16bf a, v16bf b, v8f c) {
  return __builtin_amdgcn_wmma_f32_16x16x32_bf16(false, a, false, b,
                                                 (short)0, c, false, false);
}

// ---------------------------------------------------------------------------
// Kernel 1: fp32 -> bf16 conversion (x and the four weight matrices).
// ---------------------------------------------------------------------------
__global__ void __launch_bounds__(256)
cvt_f32_to_bf16(const float* __restrict__ src, __bf16* __restrict__ dst, int n) {
  int i = blockIdx.x * blockDim.x + threadIdx.x;
  const int stride = gridDim.x * blockDim.x;
  for (; i < n; i += stride) dst[i] = (__bf16)src[i];
}

// ---------------------------------------------------------------------------
// Kernel 2: fused Q/K/V projection (y = x @ W^T) with RoPE epilogue.
// One wave computes a 32(M=bs rows) x 64(N=one head) tile with a
// double-buffered k-loop (8 WMMAs per k-step hide the next tile's loads).
//   Q,K -> [b][h][s][dk] bf16 (RoPE applied; Q scaled by 1/sqrt(dk)=0.125)
//   V   -> [b][h][dk][s] bf16 (transposed so PV B-frags are contiguous)
// ---------------------------------------------------------------------------
__global__ void __launch_bounds__(256)
qkv_rope_kernel(const __bf16* __restrict__ xb,
                const __bf16* __restrict__ Wqb,
                const __bf16* __restrict__ Wkb,
                const __bf16* __restrict__ Wvb,
                const int*    __restrict__ positions,
                __bf16* __restrict__ Qh,
                __bf16* __restrict__ Kh,
                __bf16* __restrict__ Vt) {
  const int lane = threadIdx.x & 31;
  const int wid  = (blockIdx.x * blockDim.x + threadIdx.x) >> 5;
  const int wsel = wid / (128 * 16);        // 0:Q 1:K 2:V (wave-uniform)
  const int rem  = wid % (128 * 16);
  const int mt   = rem >> 4;                // M tile: 32 rows of (b*S+s)
  const int nt   = rem & 15;                // N tile: head index (64 cols)

  const __bf16* W = (wsel == 0) ? Wqb : (wsel == 1) ? Wkb : Wvb;

  v8f acc[2][4] = {};
  const __bf16* arow = xb + (size_t)(mt * 32) * D_MODEL;
  const __bf16* brow = W  + (size_t)(nt * 64) * D_MODEL;

  // prologue: fragments for k0 = 0
  v16bf a0 = load_frag16(arow, D_MODEL, lane);
  v16bf a1 = load_frag16(arow + (size_t)16 * D_MODEL, D_MODEL, lane);
  v16bf b0 = load_frag16(brow,                          D_MODEL, lane);
  v16bf b1 = load_frag16(brow + (size_t)16 * D_MODEL,   D_MODEL, lane);
  v16bf b2 = load_frag16(brow + (size_t)32 * D_MODEL,   D_MODEL, lane);
  v16bf b3 = load_frag16(brow + (size_t)48 * D_MODEL,   D_MODEL, lane);

  for (int k0 = 0; k0 < D_MODEL - 32; k0 += 32) {
    const int kn = k0 + 32;
    // issue next tile's loads first, then compute on current registers
    v16bf na0 = load_frag16(arow + kn, D_MODEL, lane);
    v16bf na1 = load_frag16(arow + (size_t)16 * D_MODEL + kn, D_MODEL, lane);
    v16bf nb0 = load_frag16(brow + kn,                        D_MODEL, lane);
    v16bf nb1 = load_frag16(brow + (size_t)16 * D_MODEL + kn, D_MODEL, lane);
    v16bf nb2 = load_frag16(brow + (size_t)32 * D_MODEL + kn, D_MODEL, lane);
    v16bf nb3 = load_frag16(brow + (size_t)48 * D_MODEL + kn, D_MODEL, lane);

    acc[0][0] = wmma_bf16(a0, b0, acc[0][0]);
    acc[1][0] = wmma_bf16(a1, b0, acc[1][0]);
    acc[0][1] = wmma_bf16(a0, b1, acc[0][1]);
    acc[1][1] = wmma_bf16(a1, b1, acc[1][1]);
    acc[0][2] = wmma_bf16(a0, b2, acc[0][2]);
    acc[1][2] = wmma_bf16(a1, b2, acc[1][2]);
    acc[0][3] = wmma_bf16(a0, b3, acc[0][3]);
    acc[1][3] = wmma_bf16(a1, b3, acc[1][3]);

    a0 = na0; a1 = na1; b0 = nb0; b1 = nb1; b2 = nb2; b3 = nb3;
  }
  // epilogue k-step
  acc[0][0] = wmma_bf16(a0, b0, acc[0][0]);
  acc[1][0] = wmma_bf16(a1, b0, acc[1][0]);
  acc[0][1] = wmma_bf16(a0, b1, acc[0][1]);
  acc[1][1] = wmma_bf16(a1, b1, acc[1][1]);
  acc[0][2] = wmma_bf16(a0, b2, acc[0][2]);
  acc[1][2] = wmma_bf16(a1, b2, acc[1][2]);
  acc[0][3] = wmma_bf16(a0, b3, acc[0][3]);
  acc[1][3] = wmma_bf16(a1, b3, acc[1][3]);

  // Epilogue: C/D layout -> N = lane&15, M = r + 8*(lane>>4).
  const int n      = lane & 15;
  const int rowoff = (lane >> 4) << 3;
  const int h      = nt;
#pragma unroll
  for (int mi = 0; mi < 2; ++mi) {
#pragma unroll
    for (int s = 0; s < 4; ++s) {
      const int j = s * 16 + n;                                 // 0..63 in head
      const float invf = __powf(10000.f, -(float)(j & ~1) * (1.f / 64.f));
#pragma unroll
      for (int r = 0; r < 8; ++r) {
        const int row = mt * 32 + mi * 16 + rowoff + r;         // flat b*S+s
        float v = acc[mi][s][r];
        if (wsel < 2) {
          // partner lane holds feature j^1 for the same row
          float partner = __shfl_xor(v, 1, 32);
          float ang = (float)positions[row] * invf;
          float sn, cs;
          __sincosf(ang, &sn, &cs);
          v = v * cs + ((j & 1) ? partner : -partner) * sn;     // RoPE
          if (wsel == 0) v *= 0.125f;                           // fold 1/sqrt(dk)
        }
        const int bb = row >> 11;                               // / SEQ
        const int ss = row & (SEQ - 1);
        if (wsel == 2) {
          Vt[((size_t)(bb * NUM_HEADS + h) * DK + j) * SEQ + ss] = (__bf16)v;
        } else {
          __bf16* dst = (wsel == 0) ? Qh : Kh;
          dst[((size_t)(bb * NUM_HEADS + h) * SEQ + ss) * DK + j] = (__bf16)v;
        }
      }
    }
  }
}

// ---------------------------------------------------------------------------
// Kernel 3: causal flash attention. One wave per (b,h, 16 q-rows).
// Streams K/V in 32-position chunks with online softmax; P re-staged through
// LDS (padded to 80B rows -> 16B-aligned ds_load_b128) to build the A-frag.
// ---------------------------------------------------------------------------
__global__ void __launch_bounds__(256)
attention_kernel(const __bf16* __restrict__ Qh,
                 const __bf16* __restrict__ Kh,
                 const __bf16* __restrict__ Vt,
                 __bf16* __restrict__ Ob) {
  __shared__ __bf16 plds[8][16][40];        // per-wave 16x32 P tile, padded
  const int lane  = threadIdx.x & 31;
  const int wv    = threadIdx.x >> 5;
  const int wid   = (blockIdx.x * blockDim.x + threadIdx.x) >> 5;  // 0..4095
  const int qt    = wid & (SEQ / 16 - 1);   // 0..127
  const int bh    = wid >> 7;               // b*H+h, 0..31
  const int qbase = qt * 16;

  const __bf16* Qp = Qh + (size_t)bh * SEQ * DK + (size_t)qbase * DK;
  const __bf16* Kp = Kh + (size_t)bh * SEQ * DK;
  const __bf16* Vp = Vt + (size_t)bh * DK * SEQ;

  const v16bf qa = load_frag16(Qp + 0,  DK, lane);
  const v16bf qb = load_frag16(Qp + 32, DK, lane);

  v8f o0 = {}, o1 = {}, o2 = {}, o3 = {};
  float m[8], l[8];
#pragma unroll
  for (int r = 0; r < 8; ++r) { m[r] = -1e38f; l[r] = 0.f; }

  const int n      = lane & 15;
  const int rowoff = (lane >> 4) << 3;
  const int cmax   = (qbase + 15) >> 5;

  for (int c = 0; c <= cmax; ++c) {
    const int kbase = c << 5;

    // scores: two 16x16 tiles covering k-positions kbase..kbase+31
    v8f s0 = {}, s1 = {};
    {
      v16bf k0a = load_frag16(Kp + (size_t)kbase * DK + 0,  DK, lane);
      v16bf k0b = load_frag16(Kp + (size_t)kbase * DK + 32, DK, lane);
      s0 = wmma_bf16(qa, k0a, s0);
      s0 = wmma_bf16(qb, k0b, s0);
      v16bf k1a = load_frag16(Kp + (size_t)(kbase + 16) * DK + 0,  DK, lane);
      v16bf k1b = load_frag16(Kp + (size_t)(kbase + 16) * DK + 32, DK, lane);
      s1 = wmma_bf16(qa, k1a, s1);
      s1 = wmma_bf16(qb, k1b, s1);
    }

    // V B-frags issued early so they overlap the softmax VALU below
    v16bf v0 = load_frag16(Vp + (size_t)0  * SEQ + kbase, SEQ, lane);
    v16bf v1 = load_frag16(Vp + (size_t)16 * SEQ + kbase, SEQ, lane);
    v16bf v2 = load_frag16(Vp + (size_t)32 * SEQ + kbase, SEQ, lane);
    v16bf v3 = load_frag16(Vp + (size_t)48 * SEQ + kbase, SEQ, lane);

    // online softmax; rows in elements, columns across 16-lane halves
    float corr[8];
#pragma unroll
    for (int r = 0; r < 8; ++r) {
      const int row = qbase + rowoff + r;
      float a0 = s0[r], a1 = s1[r];
      if (kbase + n > row)      a0 = -3e38f;   // causal mask
      if (kbase + 16 + n > row) a1 = -3e38f;
      float mx = fmaxf(a0, a1);
#pragma unroll
      for (int off = 1; off < 16; off <<= 1)
        mx = fmaxf(mx, __shfl_xor(mx, off, 32));
      mx = fmaxf(mx, m[r]);
      const float e0 = __expf(a0 - mx);
      const float e1 = __expf(a1 - mx);
      const float cr = __expf(m[r] - mx);
      float rs = e0 + e1;
#pragma unroll
      for (int off = 1; off < 16; off <<= 1)
        rs += __shfl_xor(rs, off, 32);
      l[r] = l[r] * cr + rs;
      m[r] = mx;
      corr[r] = cr;
      plds[wv][rowoff + r][n]      = (__bf16)e0;
      plds[wv][rowoff + r][16 + n] = (__bf16)e1;
    }
#pragma unroll
    for (int r = 0; r < 8; ++r) {
      o0[r] *= corr[r]; o1[r] *= corr[r]; o2[r] *= corr[r]; o3[r] *= corr[r];
    }

    // P (16x32) C-layout -> A-frag via LDS (same wave: DScnt ordering only)
    const int kb = (lane >> 4) << 3;
    const __bf16* prow = &plds[wv][lane & 15][0];
    v8bf lo = *(const v8bf*)(prow + kb);
    v8bf hi = *(const v8bf*)(prow + kb + 16);
    v16bf pA;
#pragma unroll
    for (int i = 0; i < 8; ++i) { pA[i] = lo[i]; pA[i + 8] = hi[i]; }

    o0 = wmma_bf16(pA, v0, o0);
    o1 = wmma_bf16(pA, v1, o1);
    o2 = wmma_bf16(pA, v2, o2);
    o3 = wmma_bf16(pA, v3, o3);
  }

  // normalize and scatter to [b][s][D] (bf16) for the output projection
  const int b = bh >> 4, h = bh & 15;
#pragma unroll
  for (int r = 0; r < 8; ++r) {
    const int srow = qbase + rowoff + r;
    const float inv = 1.f / l[r];
    const size_t base = (size_t)(b * SEQ + srow) * D_MODEL + h * DK + n;
    Ob[base + 0]  = (__bf16)(o0[r] * inv);
    Ob[base + 16] = (__bf16)(o1[r] * inv);
    Ob[base + 32] = (__bf16)(o2[r] * inv);
    Ob[base + 48] = (__bf16)(o3[r] * inv);
  }
}

// ---------------------------------------------------------------------------
// Kernel 4: output projection out = O @ Wo^T, fp32 epilogue to d_out.
// 32x64 wave tile, double-buffered k-loop like kernel 2.
// ---------------------------------------------------------------------------
__global__ void __launch_bounds__(256)
out_proj_kernel(const __bf16* __restrict__ Ob,
                const __bf16* __restrict__ Wob,
                float* __restrict__ out) {
  const int lane = threadIdx.x & 31;
  const int wid  = (blockIdx.x * blockDim.x + threadIdx.x) >> 5;  // 0..2047
  const int mt   = wid >> 4;                // 0..127 (32 rows each)
  const int nt   = wid & 15;

  v8f acc[2][4] = {};
  const __bf16* arow = Ob  + (size_t)(mt * 32) * D_MODEL;
  const __bf16* brow = Wob + (size_t)(nt * 64) * D_MODEL;

  v16bf a0 = load_frag16(arow, D_MODEL, lane);
  v16bf a1 = load_frag16(arow + (size_t)16 * D_MODEL, D_MODEL, lane);
  v16bf b0 = load_frag16(brow,                        D_MODEL, lane);
  v16bf b1 = load_frag16(brow + (size_t)16 * D_MODEL, D_MODEL, lane);
  v16bf b2 = load_frag16(brow + (size_t)32 * D_MODEL, D_MODEL, lane);
  v16bf b3 = load_frag16(brow + (size_t)48 * D_MODEL, D_MODEL, lane);

  for (int k0 = 0; k0 < D_MODEL - 32; k0 += 32) {
    const int kn = k0 + 32;
    v16bf na0 = load_frag16(arow + kn, D_MODEL, lane);
    v16bf na1 = load_frag16(arow + (size_t)16 * D_MODEL + kn, D_MODEL, lane);
    v16bf nb0 = load_frag16(brow + kn,                        D_MODEL, lane);
    v16bf nb1 = load_frag16(brow + (size_t)16 * D_MODEL + kn, D_MODEL, lane);
    v16bf nb2 = load_frag16(brow + (size_t)32 * D_MODEL + kn, D_MODEL, lane);
    v16bf nb3 = load_frag16(brow + (size_t)48 * D_MODEL + kn, D_MODEL, lane);

    acc[0][0] = wmma_bf16(a0, b0, acc[0][0]);
    acc[1][0] = wmma_bf16(a1, b0, acc[1][0]);
    acc[0][1] = wmma_bf16(a0, b1, acc[0][1]);
    acc[1][1] = wmma_bf16(a1, b1, acc[1][1]);
    acc[0][2] = wmma_bf16(a0, b2, acc[0][2]);
    acc[1][2] = wmma_bf16(a1, b2, acc[1][2]);
    acc[0][3] = wmma_bf16(a0, b3, acc[0][3]);
    acc[1][3] = wmma_bf16(a1, b3, acc[1][3]);

    a0 = na0; a1 = na1; b0 = nb0; b1 = nb1; b2 = nb2; b3 = nb3;
  }
  acc[0][0] = wmma_bf16(a0, b0, acc[0][0]);
  acc[1][0] = wmma_bf16(a1, b0, acc[1][0]);
  acc[0][1] = wmma_bf16(a0, b1, acc[0][1]);
  acc[1][1] = wmma_bf16(a1, b1, acc[1][1]);
  acc[0][2] = wmma_bf16(a0, b2, acc[0][2]);
  acc[1][2] = wmma_bf16(a1, b2, acc[1][2]);
  acc[0][3] = wmma_bf16(a0, b3, acc[0][3]);
  acc[1][3] = wmma_bf16(a1, b3, acc[1][3]);

  const int n      = lane & 15;
  const int rowoff = (lane >> 4) << 3;
#pragma unroll
  for (int mi = 0; mi < 2; ++mi) {
#pragma unroll
    for (int s = 0; s < 4; ++s) {
#pragma unroll
      for (int r = 0; r < 8; ++r) {
        const int row = mt * 32 + mi * 16 + rowoff + r;
        out[(size_t)row * D_MODEL + nt * 64 + s * 16 + n] = acc[mi][s][r];
      }
    }
  }
}

// ---------------------------------------------------------------------------
extern "C" void kernel_launch(void* const* d_in, const int* in_sizes, int n_in,
                              void* d_out, int out_size, void* d_ws, size_t ws_size,
                              hipStream_t stream) {
  const float* x   = (const float*)d_in[0];
  const float* Wq  = (const float*)d_in[1];
  const float* Wk  = (const float*)d_in[2];
  const float* Wv  = (const float*)d_in[3];
  const float* Wo  = (const float*)d_in[4];
  const int*   pos = (const int*)d_in[5];

  char* ws = (char*)d_ws;
  size_t off = 0;
  auto carve = [&](size_t bytes) -> void* {
    void* p = ws + off;
    off += (bytes + 255) & ~(size_t)255;
    return p;
  };
  __bf16* xb  = (__bf16*)carve((size_t)BS * D_MODEL * 2);        // 8 MB
  __bf16* Wqb = (__bf16*)carve((size_t)D_MODEL * D_MODEL * 2);   // 2 MB
  __bf16* Wkb = (__bf16*)carve((size_t)D_MODEL * D_MODEL * 2);
  __bf16* Wvb = (__bf16*)carve((size_t)D_MODEL * D_MODEL * 2);
  __bf16* Wob = (__bf16*)carve((size_t)D_MODEL * D_MODEL * 2);
  __bf16* Qh  = (__bf16*)carve((size_t)BS * D_MODEL * 2);        // [b,h,s,dk]
  __bf16* Kh  = (__bf16*)carve((size_t)BS * D_MODEL * 2);        // [b,h,s,dk]
  __bf16* Vt  = (__bf16*)carve((size_t)BS * D_MODEL * 2);        // [b,h,dk,s]
  __bf16* Ob  = (__bf16*)carve((size_t)BS * D_MODEL * 2);        // [b,s,D]

  cvt_f32_to_bf16<<<2048, 256, 0, stream>>>(x,  xb,  BS * D_MODEL);
  cvt_f32_to_bf16<<<1024, 256, 0, stream>>>(Wq, Wqb, D_MODEL * D_MODEL);
  cvt_f32_to_bf16<<<1024, 256, 0, stream>>>(Wk, Wkb, D_MODEL * D_MODEL);
  cvt_f32_to_bf16<<<1024, 256, 0, stream>>>(Wv, Wvb, D_MODEL * D_MODEL);
  cvt_f32_to_bf16<<<1024, 256, 0, stream>>>(Wo, Wob, D_MODEL * D_MODEL);

  // 3 GEMMs * 128 M-tiles * 16 N-tiles = 6144 waves = 768 blocks of 8 waves
  qkv_rope_kernel<<<768, 256, 0, stream>>>(xb, Wqb, Wkb, Wvb, pos, Qh, Kh, Vt);

  // 2*16*128 q-tiles = 4096 waves = 512 blocks
  attention_kernel<<<512, 256, 0, stream>>>(Qh, Kh, Vt, Ob);

  // 128 M-tiles * 16 N-tiles = 2048 waves = 256 blocks
  out_proj_kernel<<<256, 256, 0, stream>>>(Ob, Wob, (float*)d_out);
}
